// RPN_5368709120147
// MI455X (gfx1250) — matrix operand order repaired and checked by reading, not compiled, not run on hardware
//
#include <hip/hip_runtime.h>
#include <hip/hip_bf16.h>
#include <math.h>

typedef __attribute__((ext_vector_type(16))) _Float16 v16h;
typedef __attribute__((ext_vector_type(8)))  _Float16 v8h;
typedef __attribute__((ext_vector_type(8)))  float    v8f;
typedef __attribute__((ext_vector_type(4)))  int      v4i;

#define NBATCH 4
#define CCH    256
#define KTOT   2304          // 9 taps * 256 channels
#define KSTEPS (KTOT / 32)
#define NMS_TH 0.7f
#define BBCLIP 4.135166556742356f   // log(1000/16)
#define IMG_HI 799.0f

#if __has_builtin(__builtin_amdgcn_global_load_async_to_lds_b128)
#define HAVE_ASYNC_LDS 1
#else
#define HAVE_ASYNC_LDS 0
#endif

// builtin signature (from hipcc diagnostic): (v4i AS1*, v4i AS3*, Imm offset, Imm cpol)
#define AS1CAST(p) ((__attribute__((address_space(1))) v4i*)(p))
#define AS3CAST(p) ((__attribute__((address_space(3))) v4i*)(p))

#if HAVE_ASYNC_LDS
#if __has_builtin(__builtin_amdgcn_s_wait_asynccnt)
#define WAIT_ASYNC(n) __builtin_amdgcn_s_wait_asynccnt(n)
#else
#define WAIT_ASYNC(n) asm volatile("s_wait_asynccnt %0" :: "i"(n))
#endif
#else
#define WAIT_ASYNC(n) do {} while (0)
#endif

__device__ __forceinline__ unsigned fkey(float f) {
  unsigned u = __float_as_uint(f);
  return u ^ ((unsigned)((int)u >> 31) | 0x80000000u);   // monotonic total order
}

// ---------------- zero pad (for uniform async halo loads); rewritten every launch ----------------
__global__ void zero_pad_kernel(_Float16* z) { z[threadIdx.x] = (_Float16)0.f; }

// ---------------- weight pack: OIHW f32 -> [n][k=(tap*256+c)] f16 ----------------
__global__ void pack_weights_kernel(const float* __restrict__ w, _Float16* __restrict__ wpk) {
  int g = blockIdx.x * blockDim.x + threadIdx.x;
  if (g >= CCH * KTOT) return;
  int n = g / KTOT, k = g % KTOT;
  int tap = k >> 8, c = k & 255;
  int ky = tap / 3, kx = tap % 3;
  wpk[g] = (_Float16)w[((n * CCH + c) * 3 + ky) * 3 + kx];
}

// ---------------- NCHW f32 -> NHWC f16 ----------------
__global__ void to_nhwc_kernel(const float* __restrict__ x, _Float16* __restrict__ y, int HW) {
  long long g = (long long)blockIdx.x * blockDim.x + threadIdx.x;
  long long tot = (long long)NBATCH * HW * CCH;
  if (g >= tot) return;
  int c = (int)(g & 255);
  long long bp = g >> 8;                 // b*HW + pix
  int pix = (int)(bp % HW);
  int b   = (int)(bp / HW);
  y[g] = (_Float16)x[((long long)(b * CCH + c)) * HW + pix];
}

// ---------------- 3x3 conv as implicit GEMM on v_wmma_f32_16x16x32_f16 ----------------
// block = 256 threads (8 waves). Tile: M=128 pixels x N=128 out-channels, K-step 32,
// double-buffered LDS fed by GLOBAL_LOAD_ASYNC_TO_LDS_B128 (ASYNCcnt) when available.
// Wave w: M sub-tile (w&3)*32, N sub-tile (w>>2)*64 -> 8 WMMAs per K-step per wave.
__launch_bounds__(256)
__global__ void conv3x3_wmma_kernel(const _Float16* __restrict__ nhwc,
                                    const _Float16* __restrict__ wpk,
                                    const float* __restrict__ bias,
                                    _Float16* __restrict__ t,
                                    const _Float16* __restrict__ zpad,
                                    int Wd, int HW, int tilesM) {
  __shared__ _Float16 As[2][128][40];   // [buf][pixel][K(+pad)]
  __shared__ _Float16 Bs[2][128][40];   // [buf][n][K(+pad)]

  const int tid  = threadIdx.x;
  const int lane = tid & 31;
  const int wave = tid >> 5;
  const int wm   = wave & 3;            // 32-row M sub-tile
  const int wn   = wave >> 2;           // 64-col N sub-tile
  const int b  = blockIdx.x / tilesM;
  const int tm = blockIdx.x % tilesM;
  const int n0 = blockIdx.y * 128;
  const int m_base = tm * 128;

  // cooperative loader: row = tid>>1 (0..127), 16-half chunk = tid&1
  const int lrow   = tid >> 1;
  const int lchunk = tid & 1;
  const int m  = m_base + lrow;
  const int py = m / Wd, px = m % Wd;
  const bool mvalid = (m < HW);

  const int e    = lane >> 4;           // lane-half: selects K-half / M-offset 8
  const int rsel = lane & 15;

  v8f acc[2][4] = {};

  auto issue = [&](int buf, int kt) {
    const int tap = kt >> 8;            // 32-chunk never straddles a tap
    const int ky = tap / 3 - 1, kx = tap % 3 - 1;
    const int c0 = (kt & 255) + lchunk * 16;
    const int sy = py + ky, sx = px + kx;

    _Float16* adst = &As[buf][lrow][lchunk * 16];
    const bool av = mvalid && sy >= 0 && sy < Wd && sx >= 0 && sx < Wd;
    const _Float16* ag = av
        ? nhwc + (((long long)b * HW + (long long)sy * Wd + sx) << 8) + c0
        : zpad;                          // uniform issue count for ASYNCcnt bookkeeping
    const _Float16* bg = wpk + (long long)(n0 + lrow) * KTOT + kt + lchunk * 16;
    _Float16* bdst = &Bs[buf][lrow][lchunk * 16];
#if HAVE_ASYNC_LDS
    __builtin_amdgcn_global_load_async_to_lds_b128(AS1CAST(ag),     AS3CAST(adst),     0, 0);
    __builtin_amdgcn_global_load_async_to_lds_b128(AS1CAST(ag + 8), AS3CAST(adst + 8), 0, 0);
    __builtin_amdgcn_global_load_async_to_lds_b128(AS1CAST(bg),     AS3CAST(bdst),     0, 0);
    __builtin_amdgcn_global_load_async_to_lds_b128(AS1CAST(bg + 8), AS3CAST(bdst + 8), 0, 0);
#else
    const float4* a4 = (const float4*)ag;
    ((float4*)adst)[0] = a4[0]; ((float4*)adst)[1] = a4[1];
    const float4* b4 = (const float4*)bg;
    ((float4*)bdst)[0] = b4[0]; ((float4*)bdst)[1] = b4[1];
#endif
  };

  issue(0, 0);
  for (int s = 0; s < KSTEPS; ++s) {
    const int cur = s & 1;
    if (s + 1 < KSTEPS) {
      issue(cur ^ 1, (s + 1) * 32);     // stream next tile while current computes
      WAIT_ASYNC(4);                    // current buffer's 4 loads retired (in-order)
    } else {
      WAIT_ASYNC(0);
    }
    __syncthreads();

    // A fragments: lane<16 -> M=lane, K {0..7,16..23}; lane>=16 -> M=lane-16, K {8..15,24..31}
    v16h afrag[2];
    #pragma unroll
    for (int ai = 0; ai < 2; ++ai) {
      const v8h* ar = (const v8h*)&As[cur][wm * 32 + ai * 16 + rsel][0];
      v8h alo = ar[e], ahi = ar[2 + e];
      #pragma unroll
      for (int i = 0; i < 8; ++i) { afrag[ai][i] = alo[i]; afrag[ai][8 + i] = ahi[i]; }
    }
    #pragma unroll
    for (int j = 0; j < 4; ++j) {
      // B fragment: lane<16 -> K {0..7,16..23}; lane>=16 -> K {8..15,24..31}, N = rsel within the 16-wide tile
      const v8h* br = (const v8h*)&Bs[cur][wn * 64 + j * 16 + rsel][0];
      v8h blo = br[e], bhi = br[2 + e];
      v16h bfrag;
      #pragma unroll
      for (int i = 0; i < 8; ++i) { bfrag[i] = blo[i]; bfrag[8 + i] = bhi[i]; }
      acc[0][j] = __builtin_amdgcn_wmma_f32_16x16x32_f16(false, afrag[0], false, bfrag,
                                                         (short)0, acc[0][j], false, false);
      acc[1][j] = __builtin_amdgcn_wmma_f32_16x16x32_f16(false, afrag[1], false, bfrag,
                                                         (short)0, acc[1][j], false, false);
    }
    __syncthreads();                    // protect buffer 'cur' before it is refilled
  }

  // epilogue: C/D layout -> VGPR r: M = r + 8*e, N = rsel; fuse bias + ReLU, store f16 NHWC
  #pragma unroll
  for (int ai = 0; ai < 2; ++ai) {
    #pragma unroll
    for (int j = 0; j < 4; ++j) {
      const int ng = n0 + wn * 64 + j * 16 + rsel;
      const float bv = bias[ng];
      #pragma unroll
      for (int r = 0; r < 8; ++r) {
        const int mg = m_base + wm * 32 + ai * 16 + r + e * 8;
        if (mg < HW) {
          float v = acc[ai][j][r] + bv;
          v = v > 0.f ? v : 0.f;
          t[(((long long)b * HW + mg) << 8) + ng] = (_Float16)v;
        }
      }
    }
  }
}

// ---------------- head: 256 -> 3 logits + 12 deltas per pixel ----------------
__launch_bounds__(256)
__global__ void head_kernel(const _Float16* __restrict__ t,
                            const float* __restrict__ cls_w, const float* __restrict__ cls_b,
                            const float* __restrict__ bbox_w, const float* __restrict__ bbox_b,
                            float* __restrict__ scores, float* __restrict__ deltas, int HW) {
  __shared__ float wsm[15][CCH];
  __shared__ float bsm[15];
  const int tid = threadIdx.x;
  #pragma unroll
  for (int r = 0; r < 15; ++r)
    wsm[r][tid] = (r < 3) ? cls_w[r * CCH + tid] : bbox_w[(r - 3) * CCH + tid];
  if (tid < 15) bsm[tid] = (tid < 3) ? cls_b[tid] : bbox_b[tid - 3];
  __syncthreads();

  const int p = blockIdx.x * 256 + tid;
  if (p >= NBATCH * HW) return;
  const int b = p / HW, pix = p % HW;
  const _Float16* row = t + (((long long)b * HW + pix) << 8);

  float acc[15];
  #pragma unroll
  for (int o = 0; o < 15; ++o) acc[o] = bsm[o];
  for (int c = 0; c < CCH; ++c) {
    const float v = (float)row[c];
    #pragma unroll
    for (int o = 0; o < 15; ++o) acc[o] += v * wsm[o][c];
  }
  const int N3 = HW * 3;
  #pragma unroll
  for (int a = 0; a < 3; ++a) {
    scores[(long long)b * N3 + pix * 3 + a] = 1.f / (1.f + expf(-acc[a]));
    #pragma unroll
    for (int d = 0; d < 4; ++d)
      deltas[(((long long)b * N3 + pix * 3 + a) << 2) + d] = acc[3 + a * 4 + d];
  }
}

// ---------------- top-k: two-round radix select (256 bins) ----------------
__launch_bounds__(256)
__global__ void topk_select_kernel(const float* __restrict__ sc, int n, int in_stride, int k,
                                   float* __restrict__ sel_s, int* __restrict__ sel_i, int out_stride) {
  const float* in = sc + (long long)blockIdx.x * in_stride;
  float* os = sel_s + (long long)blockIdx.x * out_stride;
  int*   oi = sel_i + (long long)blockIdx.x * out_stride;
  __shared__ unsigned hist[256];
  __shared__ unsigned sh_t0, sh_above0, sh_thresh, sh_cntHigh, sh_cntEq;
  const int tid = threadIdx.x;

  hist[tid] = 0; __syncthreads();
  for (int i = tid; i < n; i += 256) atomicAdd(&hist[fkey(in[i]) >> 24], 1u);
  __syncthreads();
  if (tid == 0) {
    unsigned cum = 0, above = 0; int t0 = 0;
    for (int bkt = 255; bkt >= 0; --bkt) {
      if (cum + hist[bkt] >= (unsigned)k) { t0 = bkt; above = cum; break; }
      cum += hist[bkt];
    }
    sh_t0 = (unsigned)t0; sh_above0 = above;
  }
  __syncthreads();
  const unsigned t0 = sh_t0;

  hist[tid] = 0; __syncthreads();
  for (int i = tid; i < n; i += 256) {
    unsigned key = fkey(in[i]);
    if ((key >> 24) == t0) atomicAdd(&hist[(key >> 16) & 255], 1u);
  }
  __syncthreads();
  if (tid == 0) {
    unsigned k1 = (unsigned)k - sh_above0, cum = 0, above = 0; int t1 = 0;
    for (int bkt = 255; bkt >= 0; --bkt) {
      if (cum + hist[bkt] >= k1) { t1 = bkt; above = cum; break; }
      cum += hist[bkt];
    }
    sh_thresh  = (t0 << 8) | (unsigned)t1;
    sh_cntHigh = 0;
    sh_cntEq   = sh_above0 + above;       // eq slots start after the strictly-greater ones
  }
  __syncthreads();
  const unsigned thresh = sh_thresh;
  for (int i = tid; i < n; i += 256) {
    float v = in[i];
    unsigned k16 = fkey(v) >> 16;
    if (k16 > thresh) {
      unsigned s = atomicAdd(&sh_cntHigh, 1u); os[s] = v; oi[s] = i;
    } else if (k16 == thresh) {
      unsigned s = atomicAdd(&sh_cntEq, 1u);
      if (s < (unsigned)k) { os[s] = v; oi[s] = i; }
    }
  }
  for (int j = k + tid; j < 1024; j += 256) { os[j] = -__builtin_inff(); oi[j] = -1; }
}

// ---------------- bitonic sort of 1024 (descending by score) ----------------
__launch_bounds__(256)
__global__ void sort1024_kernel(float* __restrict__ sel_s, int* __restrict__ sel_i, int seg_stride) {
  __shared__ float s[1024];
  __shared__ int   id[1024];
  float* gs = sel_s + (long long)blockIdx.x * seg_stride;
  int*   gi = sel_i + (long long)blockIdx.x * seg_stride;
  const int tid = threadIdx.x;
  for (int j = tid; j < 1024; j += 256) { s[j] = gs[j]; id[j] = gi[j]; }
  __syncthreads();
  for (int size = 2; size <= 1024; size <<= 1) {
    for (int stride = size >> 1; stride > 0; stride >>= 1) {
      for (int t = tid; t < 512; t += 256) {
        int lo = 2 * t - (t & (stride - 1));
        int hi = lo + stride;
        bool desc = ((lo & size) == 0);
        float a = s[lo], b = s[hi];
        bool sw = desc ? (a < b) : (a > b);
        if (sw) { s[lo] = b; s[hi] = a; int ti = id[lo]; id[lo] = id[hi]; id[hi] = ti; }
      }
      __syncthreads();
    }
  }
  for (int j = tid; j < 1024; j += 256) { gs[j] = s[j]; gi[j] = id[j]; }
}

// ---------------- anchor decode + clip ----------------
__global__ void decode_kernel(const int* __restrict__ sel_i, const float* __restrict__ deltas,
                              float* __restrict__ boxes, int lvl, int Wd,
                              float fsize, float fstride, int N3) {
  int g = blockIdx.x * blockDim.x + threadIdx.x;
  if (g >= NBATCH * 1000) return;
  int b = g / 1000, j = g % 1000;
  int seg = b * 5 + lvl;
  int i = sel_i[seg * 1024 + j];
  float bx0 = 0.f, bx1 = 0.f, bx2 = 0.f, bx3 = 0.f;
  if (i >= 0) {
    const float RAT[3] = {0.5f, 1.0f, 2.0f};
    int a = i % 3, p = i / 3;
    int x = p % Wd, y = p / Wd;
    float hr = sqrtf(RAT[a]);
    float wsz = fsize / hr, hsz = fsize * hr;
    float sx = x * fstride, sy = y * fstride;
    float x1 = sx - 0.5f * wsz, y1 = sy - 0.5f * hsz;
    float x2 = sx + 0.5f * wsz, y2 = sy + 0.5f * hsz;
    float wa = x2 - x1 + 1.f, ha = y2 - y1 + 1.f;
    float cx = x1 + 0.5f * wa, cy = y1 + 0.5f * ha;
    const float* d = deltas + (((long long)b * N3 + i) << 2);
    float dw = fminf(d[2], BBCLIP), dh = fminf(d[3], BBCLIP);
    float pcx = d[0] * wa + cx, pcy = d[1] * ha + cy;
    float pw = expf(dw) * wa, ph = expf(dh) * ha;
    bx0 = fminf(fmaxf(pcx - 0.5f * pw, 0.f), IMG_HI);
    bx1 = fminf(fmaxf(pcy - 0.5f * ph, 0.f), IMG_HI);
    bx2 = fminf(fmaxf(pcx + 0.5f * pw - 1.f, 0.f), IMG_HI);
    bx3 = fminf(fmaxf(pcy + 0.5f * ph - 1.f, 0.f), IMG_HI);
  }
  float* o = boxes + (((long long)(seg * 1000 + j)) << 2);
  o[0] = bx0; o[1] = bx1; o[2] = bx2; o[3] = bx3;
}

// ---------------- greedy NMS (boxes pre-sorted desc); suppressed -> -inf ----------------
__launch_bounds__(256)
__global__ void nms_kernel(float* __restrict__ sel_s, const float* __restrict__ boxes) {
  const int seg = blockIdx.x;            // b*5 + lvl
  __shared__ float bx[1000][4];
  __shared__ unsigned char keep[1000];
  const int tid = threadIdx.x;
  for (int j = tid; j < 1000; j += 256) {
    const float* p = boxes + (((long long)(seg * 1000 + j)) << 2);
    bx[j][0] = p[0]; bx[j][1] = p[1]; bx[j][2] = p[2]; bx[j][3] = p[3];
    keep[j] = (sel_s[seg * 1024 + j] > -1e37f) ? 1 : 0;
  }
  __syncthreads();
  for (int i = 0; i < 1000; ++i) {
    if (keep[i]) {
      float ix1 = bx[i][0], iy1 = bx[i][1], ix2 = bx[i][2], iy2 = bx[i][3];
      float ai = (ix2 - ix1 + 1.f) * (iy2 - iy1 + 1.f);
      for (int j = i + 1 + tid; j < 1000; j += 256) {
        if (!keep[j]) continue;
        float iw = fminf(ix2, bx[j][2]) - fmaxf(ix1, bx[j][0]) + 1.f;
        float ih = fminf(iy2, bx[j][3]) - fmaxf(iy1, bx[j][1]) + 1.f;
        iw = fmaxf(iw, 0.f); ih = fmaxf(ih, 0.f);
        float inter = iw * ih;
        float aj = (bx[j][2] - bx[j][0] + 1.f) * (bx[j][3] - bx[j][1] + 1.f);
        if (inter / (ai + aj - inter) > NMS_TH) keep[j] = 0;
      }
    }
    __syncthreads();
  }
  for (int j = tid; j < 1000; j += 256)
    if (!keep[j]) sel_s[seg * 1024 + j] = -__builtin_inff();
}

// ---------------- concat 5 levels -> per-batch 5120 (pad -inf) ----------------
__global__ void concat_kernel(const float* __restrict__ sel_s, float* __restrict__ cat_s) {
  int g = blockIdx.x * blockDim.x + threadIdx.x;
  if (g >= NBATCH * 5120) return;
  int b = g / 5120, c = g % 5120;
  float v = -__builtin_inff();
  if (c < 5000) { int lvl = c / 1000, j = c % 1000; v = sel_s[(b * 5 + lvl) * 1024 + j]; }
  cat_s[g] = v;
}

// ---------------- final gather: d_out = final_b [4,1000,4] ++ final_s [4,1000] ----------------
__global__ void writeout_kernel(const float* __restrict__ sel2_s, const int* __restrict__ sel2_i,
                                const float* __restrict__ boxes, float* __restrict__ out) {
  int g = blockIdx.x * blockDim.x + threadIdx.x;
  if (g >= NBATCH * 1000) return;
  int b = g / 1000, j = g % 1000;
  float s = sel2_s[b * 1024 + j];
  int i = sel2_i[b * 1024 + j];
  float b0 = 0.f, b1 = 0.f, b2 = 0.f, b3 = 0.f;
  if (i >= 0 && i < 5000) {
    int lvl = i / 1000, jj = i % 1000;
    const float* p = boxes + (((long long)((b * 5 + lvl) * 1000 + jj)) << 2);
    b0 = p[0]; b1 = p[1]; b2 = p[2]; b3 = p[3];
  }
  float* ob = out + (((long long)(b * 1000 + j)) << 2);
  ob[0] = b0; ob[1] = b1; ob[2] = b2; ob[3] = b3;
  out[16000 + b * 1000 + j] = s;
}

// =======================================================================================
extern "C" void kernel_launch(void* const* d_in, const int* in_sizes, int n_in,
                              void* d_out, int out_size, void* d_ws, size_t ws_size,
                              hipStream_t stream) {
  const float* feats[5];
  for (int i = 0; i < 5; ++i) feats[i] = (const float*)d_in[i];
  const float* conv_w = (const float*)d_in[5];
  const float* conv_b = (const float*)d_in[6];
  const float* cls_w  = (const float*)d_in[7];
  const float* cls_b  = (const float*)d_in[8];
  const float* bbox_w = (const float*)d_in[9];
  const float* bbox_b = (const float*)d_in[10];

  static const int   HWD[5] = {200, 100, 50, 25, 13};
  static const float SZ[5]  = {32.f, 64.f, 128.f, 256.f, 512.f};
  static const float STq[5] = {4.f, 8.f, 16.f, 32.f, 64.f};

  size_t off = 0;
  char* base = (char*)d_ws;
  auto alloc = [&](size_t bytes) -> void* {
    void* p = base + off;
    off = (off + bytes + 255) & ~(size_t)255;
    return p;
  };
  _Float16* wpk    = (_Float16*)alloc((size_t)CCH * KTOT * 2);
  _Float16* zpad   = (_Float16*)alloc(256);
  _Float16* nhwc   = (_Float16*)alloc((size_t)NBATCH * 40000 * CCH * 2);   // reused per level
  _Float16* tbuf   = (_Float16*)alloc((size_t)NBATCH * 40000 * CCH * 2);   // reused per level
  float*    scores = (float*)alloc((size_t)NBATCH * 120000 * 4);           // reused per level
  float*    deltas = (float*)alloc((size_t)NBATCH * 120000 * 4 * 4);       // reused per level
  float*    sel_s  = (float*)alloc((size_t)20 * 1024 * 4);
  int*      sel_i  = (int*)alloc((size_t)20 * 1024 * 4);
  float*    boxes  = (float*)alloc((size_t)20 * 1000 * 4 * 4);
  float*    cat_s  = (float*)alloc((size_t)NBATCH * 5120 * 4);
  float*    sel2_s = (float*)alloc((size_t)NBATCH * 1024 * 4);
  int*      sel2_i = (int*)alloc((size_t)NBATCH * 1024 * 4);
  (void)ws_size; (void)in_sizes; (void)n_in; (void)out_size;

  zero_pad_kernel<<<1, 128, 0, stream>>>(zpad);
  pack_weights_kernel<<<(CCH * KTOT + 255) / 256, 256, 0, stream>>>(conv_w, wpk);

  for (int lvl = 0; lvl < 5; ++lvl) {
    const int Wd = HWD[lvl], HW = Wd * Wd, N3 = HW * 3;
    const int k = (N3 < 1000) ? N3 : 1000;
    const int tilesM = (HW + 127) / 128;

    long long nElem = (long long)NBATCH * HW * CCH;
    to_nhwc_kernel<<<(unsigned)((nElem + 255) / 256), 256, 0, stream>>>(feats[lvl], nhwc, HW);

    conv3x3_wmma_kernel<<<dim3(NBATCH * tilesM, 2), 256, 0, stream>>>(
        nhwc, wpk, conv_b, tbuf, zpad, Wd, HW, tilesM);

    head_kernel<<<(NBATCH * HW + 255) / 256, 256, 0, stream>>>(
        tbuf, cls_w, cls_b, bbox_w, bbox_b, scores, deltas, HW);

    topk_select_kernel<<<NBATCH, 256, 0, stream>>>(
        scores, N3, N3, k, sel_s + lvl * 1024, sel_i + lvl * 1024, 5 * 1024);

    sort1024_kernel<<<NBATCH, 256, 0, stream>>>(
        sel_s + lvl * 1024, sel_i + lvl * 1024, 5 * 1024);

    decode_kernel<<<(NBATCH * 1000 + 255) / 256, 256, 0, stream>>>(
        sel_i, deltas, boxes, lvl, Wd, SZ[lvl], STq[lvl], N3);
  }

  nms_kernel<<<20, 256, 0, stream>>>(sel_s, boxes);
  concat_kernel<<<(NBATCH * 5120 + 255) / 256, 256, 0, stream>>>(sel_s, cat_s);
  topk_select_kernel<<<NBATCH, 256, 0, stream>>>(cat_s, 5120, 5120, 1000, sel2_s, sel2_i, 1024);
  sort1024_kernel<<<NBATCH, 256, 0, stream>>>(sel2_s, sel2_i, 1024);
  writeout_kernel<<<(NBATCH * 1000 + 255) / 256, 256, 0, stream>>>(
      sel2_s, sel2_i, boxes, (float*)d_out);
}
